// DevConv_7121055777147
// MI455X (gfx1250) — matrix operand order repaired and checked by reading, not compiled, not run on hardware
//
#include <hip/hip_runtime.h>

#define IN_CH  128
#define OUT_CH 128
#define WAVES  4            // waves per block (wave32)
#define BLOCK  (WAVES * 32)
#define TILE_E (WAVES * 16) // edges per block in kernel 1
#define WELEMS (OUT_CH * IN_CH)

typedef __attribute__((ext_vector_type(16))) __bf16 v16bf;
typedef __attribute__((ext_vector_type(8)))  __bf16 v8bf;
typedef __attribute__((ext_vector_type(4)))  __bf16 v4bf;
typedef __attribute__((ext_vector_type(8)))  float  v8f;

// ---- monotone float <-> unsigned key (so atomicMax(u32) == float max) ----
__device__ __forceinline__ unsigned fkey(float f) {
    unsigned u = __float_as_uint(f);
    return (u & 0x80000000u) ? ~u : (u | 0x80000000u);
}
__device__ __forceinline__ float kdec(unsigned k) {
    unsigned u = (k & 0x80000000u) ? (k ^ 0x80000000u) : ~k;
    return __uint_as_float(u);
}
#define EMPTY_KEY 0x007FFFFFu   // fkey(-inf)

__device__ __forceinline__ __bf16 f2bf(float f) { return (__bf16)f; }

// 16B async global->LDS copy (ASYNCcnt-tracked, no VGPR data round-trip)
__device__ __forceinline__ void async_lds_b128(void* lds_ptr, const void* gaddr) {
    unsigned lds_off = (unsigned)(uintptr_t)lds_ptr;   // low 32 bits = LDS offset
    asm volatile("global_load_async_to_lds_b128 %0, %1, off"
                 :: "v"(lds_off), "v"(gaddr) : "memory");
}
__device__ __forceinline__ void wait_async_all() {
    asm volatile("s_wait_asynccnt 0x0" ::: "memory");
}

// ---------------- init scratch keys to fkey(-inf) ----------------
__global__ void devconv_init_kernel(unsigned* __restrict__ p, long n) {
    long i = (long)blockIdx.x * blockDim.x + threadIdx.x;
    if (i < n) p[i] = EMPTY_KEY;
}

// ---------------- one-shot weight conversion f32 -> bf16 ----------------
__global__ void devconv_wconv_kernel(const float* __restrict__ Wt,
                                     const float* __restrict__ Wp,
                                     __bf16* __restrict__ wtb,
                                     __bf16* __restrict__ wpb) {
    int j = (blockIdx.x * blockDim.x + threadIdx.x) * 4;
    if (j < WELEMS) {
        float4 a = *(const float4*)(Wt + j);
        *(v4bf*)&wtb[j] = (v4bf){ f2bf(a.x), f2bf(a.y), f2bf(a.z), f2bf(a.w) };
        float4 b = *(const float4*)(Wp + j);
        *(v4bf*)&wpb[j] = (v4bf){ f2bf(b.x), f2bf(b.y), f2bf(b.z), f2bf(b.w) };
    }
}

// ---------------- kernel 1: edge GEMM + fused atomic segment-max ----------------
__global__ __launch_bounds__(BLOCK) void devconv_edge_kernel(
    const float*  __restrict__ x,
    const int*    __restrict__ erow,
    const int*    __restrict__ ecol,
    const __bf16* __restrict__ wtb,    // [OUT_CH][IN_CH] bf16, row-major
    unsigned*     __restrict__ aggr,   // [N][OUT_CH] keys
    int E)
{
    __shared__ __bf16 ldsW[WELEMS];                // 32 KB: ldsW[n*128+k] = Wt[n][k]
    __shared__ __bf16 ldsD[WAVES][16 * IN_CH];     // 16 KB: per-wave diff tile
    __shared__ int    ldsRow[TILE_E];
    __shared__ int    ldsCol[TILE_E];

    const int tid  = threadIdx.x;
    const int lane = tid & 31;
    const int w    = tid >> 5;
    const int ebase = blockIdx.x * TILE_E;
    const bool allvalid = (ebase + TILE_E) <= E;   // uniform across block

    // stage bf16 W_theta via async global->LDS (16 x b128 per thread)
#pragma unroll
    for (int i = tid * 8; i < WELEMS; i += BLOCK * 8)
        async_lds_b128(&ldsW[i], wtb + i);

    if (tid < TILE_E) {
        int e = ebase + tid;
        ldsRow[tid] = (e < E) ? erow[e] : -1;
        ldsCol[tid] = (e < E) ? ecol[e] : -1;
    }
    wait_async_all();
    __syncthreads();

    // stage this wave's 16-edge diff tile (x[row]-x[col]) as bf16
    const int em0 = w * 16;
    for (int i = lane; i < 16 * 32; i += 32) {     // 32 float4 per 128-ch row
        int m  = i >> 5;
        int c4 = (i & 31) * 4;
        int r  = ldsRow[em0 + m];
        int c  = ldsCol[em0 + m];
        float4 xr = make_float4(0.f, 0.f, 0.f, 0.f);
        float4 xc = make_float4(0.f, 0.f, 0.f, 0.f);
        if (r >= 0) {
            xr = *(const float4*)(x + (size_t)r * IN_CH + c4);
            xc = *(const float4*)(x + (size_t)c * IN_CH + c4);
        }
        v4bf d = { f2bf(xr.x - xc.x), f2bf(xr.y - xc.y),
                   f2bf(xr.z - xc.z), f2bf(xr.w - xc.w) };
        *(v4bf*)&ldsD[w][m * IN_CH + c4] = d;
    }
    __syncthreads();

    const int half = lane >> 4;     // ISA: lanes 0-15 vs 16-31
    const int mrow = lane & 15;     // A-matrix M = lane%16
    const int ncol = lane & 15;     // B/C/D N   = lane%16

    // A fragments (16x32 bf16 each), ISA layout: k = (j>>3)*16 + half*8 + (j&7)
    v16bf afrag[4];
#pragma unroll
    for (int kk = 0; kk < 4; ++kk) {
        const __bf16* base = &ldsD[w][mrow * IN_CH + kk * 32 + half * 8];
        v8bf lo = *(const v8bf*)(base);       // k = kbase+half*8   .. +7
        v8bf hi = *(const v8bf*)(base + 16);  // k = kbase+16+half*8.. +7
        v16bf a;
#pragma unroll
        for (int j = 0; j < 8; ++j) { a[j] = lo[j]; a[j + 8] = hi[j]; }
        afrag[kk] = a;
    }

    // D layout: VGPR r8 holds M = r8 + 8*half, N = lane%16.
    // Hoist row indices and precompute atomic base addresses.
    int4 n0 = *(const int4*)&ldsRow[em0 + 8 * half];
    int4 n1 = *(const int4*)&ldsRow[em0 + 8 * half + 4];
    int nodeIdx[8] = { n0.x, n0.y, n0.z, n0.w, n1.x, n1.y, n1.z, n1.w };
    unsigned* addr[8];
#pragma unroll
    for (int r8 = 0; r8 < 8; ++r8) {
        int nclamp = nodeIdx[r8] < 0 ? 0 : nodeIdx[r8];
        addr[r8] = aggr + (size_t)nclamp * OUT_CH + ncol;
    }

#pragma unroll
    for (int nt = 0; nt < 8; ++nt) {          // 8 tiles of 16 out channels
        v8f acc = {};
#pragma unroll
        for (int kk = 0; kk < 4; ++kk) {
            // B layout: k = kbase + (j>>3)*16 + half*8 + (j&7)
            const __bf16* bb = &ldsW[(nt * 16 + ncol) * IN_CH + kk * 32 + half * 8];
            v8bf l0 = *(const v8bf*)(bb);
            v8bf l1 = *(const v8bf*)(bb + 16);
            v16bf b;
#pragma unroll
            for (int j = 0; j < 8; ++j) { b[j] = l0[j]; b[j + 8] = l1[j]; }
            acc = __builtin_amdgcn_wmma_f32_16x16x32_bf16(
                false, afrag[kk], false, b, (short)0, acc, false, false);
        }
        if (allvalid) {                        // uniform scalar branch (hot path)
#pragma unroll
            for (int r8 = 0; r8 < 8; ++r8)
                atomicMax(addr[r8] + nt * 16, fkey(acc[r8]));
        } else {                               // tail blocks only
#pragma unroll
            for (int r8 = 0; r8 < 8; ++r8)
                if (nodeIdx[r8] >= 0)
                    atomicMax(addr[r8] + nt * 16, fkey(acc[r8]));
        }
    }
}

// ---------------- kernel 2: decode segment-max, node GEMM with W_phi ----------------
__global__ __launch_bounds__(BLOCK) void devconv_node_kernel(
    const unsigned* __restrict__ aggr,
    const __bf16*   __restrict__ wpb,    // [OUT_CH][OUT_CH] bf16, row-major
    float*          __restrict__ out,    // [N][OUT_CH]
    int N)
{
    __shared__ __bf16 ldsW[OUT_CH * OUT_CH];    // 32 KB
    __shared__ __bf16 ldsA[WAVES][16 * OUT_CH]; // 16 KB

    const int tid  = threadIdx.x;
    const int lane = tid & 31;
    const int w    = tid >> 5;

#pragma unroll
    for (int i = tid * 8; i < OUT_CH * OUT_CH; i += BLOCK * 8)
        async_lds_b128(&ldsW[i], wpb + i);

    const int ntiles = (N + 15) / 16;
    const int tile   = blockIdx.x * WAVES + w;
    const bool active = (tile < ntiles);
    const int nbase  = tile * 16;
    const bool fulltile = active && (nbase + 16 <= N);   // uniform per wave

    if (active) {
        for (int i = lane; i < 16 * 32; i += 32) {
            int m  = i >> 5;
            int c4 = (i & 31) * 4;
            int node = nbase + m;
            v4bf d = { (__bf16)0.f, (__bf16)0.f, (__bf16)0.f, (__bf16)0.f };
            if (node < N) {
                uint4 kv = *(const uint4*)(aggr + (size_t)node * OUT_CH + c4);
                float f0 = kdec(kv.x), f1 = kdec(kv.y), f2 = kdec(kv.z), f3 = kdec(kv.w);
                // empty segments (-inf) -> 0, per reference
                if (f0 == -__builtin_huge_valf()) f0 = 0.f;
                if (f1 == -__builtin_huge_valf()) f1 = 0.f;
                if (f2 == -__builtin_huge_valf()) f2 = 0.f;
                if (f3 == -__builtin_huge_valf()) f3 = 0.f;
                d = (v4bf){ f2bf(f0), f2bf(f1), f2bf(f2), f2bf(f3) };
            }
            *(v4bf*)&ldsA[w][m * OUT_CH + c4] = d;
        }
    }
    wait_async_all();
    __syncthreads();
    if (!active) return;

    const int half = lane >> 4;
    const int mrow = lane & 15;
    const int ncol = lane & 15;

    v16bf afrag[4];
#pragma unroll
    for (int kk = 0; kk < 4; ++kk) {
        const __bf16* base = &ldsA[w][mrow * OUT_CH + kk * 32 + half * 8];
        v8bf lo = *(const v8bf*)(base);
        v8bf hi = *(const v8bf*)(base + 16);
        v16bf a;
#pragma unroll
        for (int j = 0; j < 8; ++j) { a[j] = lo[j]; a[j + 8] = hi[j]; }
        afrag[kk] = a;
    }

    // one store base per lane; all stores use constant byte offsets
    float* obase = out + (size_t)(nbase + 8 * half) * OUT_CH + ncol;

#pragma unroll
    for (int nt = 0; nt < 8; ++nt) {
        v8f acc = {};
#pragma unroll
        for (int kk = 0; kk < 4; ++kk) {
            const __bf16* bb = &ldsW[(nt * 16 + ncol) * OUT_CH + kk * 32 + half * 8];
            v8bf l0 = *(const v8bf*)(bb);
            v8bf l1 = *(const v8bf*)(bb + 16);
            v16bf b;
#pragma unroll
            for (int j = 0; j < 8; ++j) { b[j] = l0[j]; b[j + 8] = l1[j]; }
            acc = __builtin_amdgcn_wmma_f32_16x16x32_bf16(
                false, afrag[kk], false, b, (short)0, acc, false, false);
        }
        if (fulltile) {
#pragma unroll
            for (int r8 = 0; r8 < 8; ++r8)
                obase[(size_t)r8 * OUT_CH + nt * 16] = acc[r8];
        } else {
#pragma unroll
            for (int r8 = 0; r8 < 8; ++r8)
                if (nbase + r8 + 8 * half < N)
                    obase[(size_t)r8 * OUT_CH + nt * 16] = acc[r8];
        }
    }
}

extern "C" void kernel_launch(void* const* d_in, const int* in_sizes, int n_in,
                              void* d_out, int out_size, void* d_ws, size_t ws_size,
                              hipStream_t stream) {
    const float* x    = (const float*)d_in[0];
    const int*   eidx = (const int*)  d_in[1];
    const float* Wt   = (const float*)d_in[2];
    const float* Wp   = (const float*)d_in[3];

    const int N = in_sizes[0] / IN_CH;
    const int E = in_sizes[1] / 2;

    // ws layout: [N*OUT_CH u32 keys][Wt bf16 16K][Wp bf16 16K]
    unsigned* aggr = (unsigned*)d_ws;
    const long aggrN = (long)N * OUT_CH;
    __bf16* wtb = (__bf16*)((char*)d_ws + (size_t)aggrN * sizeof(unsigned));
    __bf16* wpb = wtb + WELEMS;

    devconv_init_kernel<<<(unsigned)((aggrN + 255) / 256), 256, 0, stream>>>(aggr, aggrN);
    devconv_wconv_kernel<<<(WELEMS / 4 + 255) / 256, 256, 0, stream>>>(Wt, Wp, wtb, wpb);

    const int eblocks = (E + TILE_E - 1) / TILE_E;
    devconv_edge_kernel<<<eblocks, BLOCK, 0, stream>>>(x, eidx, eidx + E, wtb, aggr, E);

    const int ntiles  = (N + 15) / 16;
    const int nblocks = (ntiles + WAVES - 1) / WAVES;
    devconv_node_kernel<<<nblocks, BLOCK, 0, stream>>>(aggr, wpb, (float*)d_out, N);
}
